// My_EncoderLayer_9672266350699
// MI455X (gfx1250) — matrix operand order repaired and checked by reading, not compiled, not run on hardware
//
#include <hip/hip_runtime.h>
#include <hip/hip_fp16.h>
#include <stdint.h>

typedef __attribute__((ext_vector_type(16))) _Float16 v16h;
typedef __attribute__((ext_vector_type(8)))  _Float16 v8h;
typedef __attribute__((ext_vector_type(8)))  float    v8f;
typedef __attribute__((ext_vector_type(4)))  float    vf4;   // 16-byte copy unit
typedef __attribute__((ext_vector_type(4)))  unsigned int u32x4;
typedef __attribute__((ext_vector_type(4)))  int      i32x4;
typedef __attribute__((ext_vector_type(8)))  int      i32x8;

union Frag { v16h v; v8h h[2]; };

// Tensor Data Mover dispatch:
//   - clang with therock headers (amd_gfx1250_TDM.h present) -> 6-arg builtin
//   - ROCm 7.2-style clang (no TDM header)                   -> 5-arg builtin
//   - builtin absent                                          -> staged fallback
#if defined(__HIP_DEVICE_COMPILE__) && __has_builtin(__builtin_amdgcn_tensor_load_to_lds)
#if __has_include(<hip/amd_detail/amd_gfx1250_TDM.h>)
#define TDM_ARITY 6
#else
#define TDM_ARITY 5
#endif
#else
#define TDM_ARITY 0
#endif

// Problem constants (from the reference)
#define NB   4
#define NC   2048
#define TOK  (NB * NC)     // 8192 tokens
#define LW   96            // L windows
#define HW   8             // H window width
#define DM   768           // D = L*H
#define DI   3072          // D_INNER

// ---------------------------------------------------------------------------
// Kernel 1: windowed attention + residual + LayerNorm1.  One block per token.
// scores[j] = q[j] * (mask[j,:] . k) / max(||q||*||k||, 1e-12)   (since
// ||q k^T||_F = ||q||*||k||), then softmax over L, res = attn*v, x = res+heads,
// LN1.  Writes x as f32 (residual) and f16 (GEMM input).
// ---------------------------------------------------------------------------
__global__ __launch_bounds__(128) void attn_ln1_kernel(
    const float* __restrict__ enc, const float* __restrict__ mask,
    const float* __restrict__ wq, const float* __restrict__ bq,
    const float* __restrict__ wk, const float* __restrict__ bk,
    const float* __restrict__ wv, const float* __restrict__ bv,
    const float* __restrict__ g1, const float* __restrict__ b1,
    float* __restrict__ Xf32, _Float16* __restrict__ Xf16)
{
    const int token = blockIdx.x;
    const int tid   = threadIdx.x;   // 0..127, threads 0..95 own a window

    __shared__ float s_k[LW], s_v[LW];
    __shared__ float s_a[128], s_b[128];

    const float* row = enc + (size_t)token * DM;

    float h8[HW];
    float q = 0.f, k = 0.f, v = 0.f;
    if (tid < LW) {
        #pragma unroll
        for (int i = 0; i < HW; i++) {
            h8[i] = row[tid * HW + i];
            q += h8[i] * wq[i];
            k += h8[i] * wk[i];
            v += h8[i] * wv[i];
        }
        q += bq[0]; k += bk[0]; v += bv[0];
        s_k[tid] = k; s_v[tid] = v;
    }
    __syncthreads();

    // ||q||^2, ||k||^2 reductions
    s_a[tid] = (tid < LW) ? q * q : 0.f;
    s_b[tid] = (tid < LW) ? k * k : 0.f;
    __syncthreads();
    for (int s = 64; s > 0; s >>= 1) {
        if (tid < s) { s_a[tid] += s_a[tid + s]; s_b[tid] += s_b[tid + s]; }
        __syncthreads();
    }
    const float nrm = fmaxf(sqrtf(s_a[0]) * sqrtf(s_b[0]), 1e-12f);
    __syncthreads();

    // masked score
    float score = 0.f;
    if (tid < LW) {
        const float* mrow = mask + tid * LW;
        float accm = 0.f;
        #pragma unroll 8
        for (int j = 0; j < LW; j++) accm += mrow[j] * s_k[j];
        score = q * accm / nrm;
    }

    // softmax over LW
    s_a[tid] = (tid < LW) ? score : -1e30f;
    __syncthreads();
    for (int s = 64; s > 0; s >>= 1) {
        if (tid < s) s_a[tid] = fmaxf(s_a[tid], s_a[tid + s]);
        __syncthreads();
    }
    const float mx = s_a[0];
    __syncthreads();
    float e = (tid < LW) ? __expf(score - mx) : 0.f;
    s_a[tid] = e;
    __syncthreads();
    for (int s = 64; s > 0; s >>= 1) {
        if (tid < s) s_a[tid] += s_a[tid + s];
        __syncthreads();
    }
    const float denom = s_a[0];
    __syncthreads();

    // residual + LN1 over DM
    float x8[HW];
    float lsum = 0.f, lsq = 0.f;
    if (tid < LW) {
        const float res = (e / denom) * s_v[tid];
        #pragma unroll
        for (int i = 0; i < HW; i++) {
            x8[i] = res + h8[i];
            lsum += x8[i];
            lsq  += x8[i] * x8[i];
        }
    }
    s_a[tid] = (tid < LW) ? lsum : 0.f;
    s_b[tid] = (tid < LW) ? lsq  : 0.f;
    __syncthreads();
    for (int s = 64; s > 0; s >>= 1) {
        if (tid < s) { s_a[tid] += s_a[tid + s]; s_b[tid] += s_b[tid + s]; }
        __syncthreads();
    }
    const float mean = s_a[0] * (1.f / DM);
    const float var  = s_b[0] * (1.f / DM) - mean * mean;
    const float inv  = rsqrtf(var + 1e-6f);

    if (tid < LW) {
        #pragma unroll
        for (int i = 0; i < HW; i++) {
            const int idx = tid * HW + i;
            const float xn = (x8[i] - mean) * inv * g1[idx] + b1[idx];
            Xf32[(size_t)token * DM + idx] = xn;
            Xf16[(size_t)token * DM + idx] = (_Float16)xn;
        }
    }
}

// ---------------------------------------------------------------------------
// Kernel 2: transpose + f32 -> f16 convert.  out[c*R + r] = (f16) in[r*C + c]
// ---------------------------------------------------------------------------
__global__ __launch_bounds__(256) void cvt_transpose_kernel(
    const float* __restrict__ in, _Float16* __restrict__ out, int R, int C)
{
    const size_t i = (size_t)blockIdx.x * blockDim.x + threadIdx.x;
    const size_t total = (size_t)R * C;
    if (i >= total) return;
    const int r = (int)(i / C);
    const int c = (int)(i % C);
    out[(size_t)c * R + r] = (_Float16)in[i];
}

// ---------------------------------------------------------------------------
// Kernel 3/4: WMMA f16 GEMM.  C = A (MxK) * B (KxN) with B given transposed
// (BT is N x K row-major, so both fragment loads are row-contiguous).
//
// Block tile 128x256, BK=32, 8 waves in a 2x4 grid; each wave owns a 64x64
// sub-tile as 4x4 v_wmma_f32_16x16x32_f16 accumulators (16 WMMA per wave per
// K-step).  LDS tiles are double-buffered; the next K-slab is moved by the
// Tensor Data Mover (one 2D D# descriptor per tile, LDS row padding 64B+16B
// reproducing the 80-byte padded LDS stride) while the matrix pipes consume
// the current slab.  Fallback without TDM: register-staged global->LDS copy.
//
// mode==1: out_f16 = relu(acc + bias[col]) ; mode==0: out_f32 = acc
// ---------------------------------------------------------------------------
#define BM   128
#define BNR  256
#define BK   32
#define LDSS 40   // padded half-stride (64B data + 16B pad per row)

#if TDM_ARITY > 0
// Issue one TDM 2D tile load: rows x 32 f16 elements from a row-major matrix
// with row stride K (elements) into LDS at lds_byte_off, padding 16B per 64B.
__device__ __forceinline__ void tdm_load_tile(const _Float16* gsrc,
                                              uint32_t lds_byte_off,
                                              int K, int rows)
{
    const uint64_t ga = (uint64_t)(uintptr_t)gsrc;
    u32x4 g0;
    g0[0] = 1u;                                   // count=1, no gather
    g0[1] = lds_byte_off;                         // lds_addr
    g0[2] = (uint32_t)ga;                         // global_addr[31:0]
    g0[3] = (uint32_t)((ga >> 32) & 0x01FFFFFFu)  // global_addr[56:32]
          | (2u << 30);                           // type=2 (image)
    i32x8 g1;
    g1[0] = (int)((1u << 16)      // data_size = 2 bytes
                | (1u << 20)      // pad_enable
                | (3u << 22)      // pad_interval: 16 DWORDs (64B)
                | (3u << 25));    // pad_amount:   4 DWORDs (16B)
    g1[1] = (int)(((uint32_t)K & 0xFFFFu) << 16);                 // tensor_dim0 lo
    g1[2] = (int)(((uint32_t)K >> 16) | (((uint32_t)rows & 0xFFFFu) << 16)); // dim0 hi | dim1 lo
    g1[3] = (int)(((uint32_t)rows >> 16) | (32u << 16));          // dim1 hi | tile_dim0=32
    g1[4] = (int)((uint32_t)rows & 0xFFFFu);                      // tile_dim1=rows, tile_dim2=0
    g1[5] = K;                                                    // tensor_dim0_stride lo
    g1[6] = 0;
    g1[7] = 0;
    i32x4 z4 = {};
#if TDM_ARITY == 5
    __builtin_amdgcn_tensor_load_to_lds(g0, g1, z4, z4, 0);
#else
    i32x8 z8 = {};
    __builtin_amdgcn_tensor_load_to_lds(g0, g1, z4, z4, z8, 0);
#endif
}
#endif

__global__ __launch_bounds__(256) void wmma_gemm_kernel(
    const _Float16* __restrict__ A,
    const _Float16* __restrict__ BT,
    float* __restrict__ outF32,
    _Float16* __restrict__ outF16,
    const float* __restrict__ bias,
    int M, int N, int K, int mode)
{
    __shared__ __align__(16) _Float16 As[2][BM][LDSS];
    __shared__ __align__(16) _Float16 Bs[2][BNR][LDSS];

    const int tid  = threadIdx.x;
    const int wave = tid >> 5;
    const int lane = tid & 31;
    const int wm = wave >> 2;        // 0..1  (64 rows each)
    const int wn = wave & 3;         // 0..3  (64 cols each)
    const int hs = (lane >> 4) & 1;  // lane half-select
    const int lr = lane & 15;

    const int blockM = blockIdx.y * BM;
    const int blockN = blockIdx.x * BNR;

    v8f acc[4][4] = {};

    const int nsteps = K / BK;

#if TDM_ARITY > 0
    const uint32_t aBase[2] = { (uint32_t)(uintptr_t)&As[0][0][0],
                                (uint32_t)(uintptr_t)&As[1][0][0] };
    const uint32_t bBase[2] = { (uint32_t)(uintptr_t)&Bs[0][0][0],
                                (uint32_t)(uintptr_t)&Bs[1][0][0] };
    const _Float16* gAt = A  + (size_t)blockM * K;   // tile origin, row 0
    const _Float16* gBt = BT + (size_t)blockN * K;

    // prologue: DMA K-slab 0 into buffer 0 (wave 0 issues; TENSORcnt tracks)
    if (tid < 32) {
        tdm_load_tile(gAt, aBase[0], K, BM);
        tdm_load_tile(gBt, bBase[0], K, BNR);
        __builtin_amdgcn_s_wait_tensorcnt(0);
    }
    __syncthreads();
#else
    // cooperative staging: per thread, chunk col ak (8 halves), base row rA.
    // A rows: rA, rA+64.  B rows: rA + 64*j, j=0..3.
    const int rA = tid >> 2;
    const int ak = (tid & 3) * 8;
    const _Float16* gA = A  + (size_t)(blockM + rA) * K + ak;
    const _Float16* gB = BT + (size_t)(blockN + rA) * K + ak;
    const size_t rs64 = (size_t)64 * K;   // 64-row stride in elements

    // prologue: load K-slab 0 straight into buffer 0
    *(vf4*)&As[0][rA     ][ak] = *(const vf4*)&gA[0];
    *(vf4*)&As[0][rA + 64][ak] = *(const vf4*)&gA[rs64];
    #pragma unroll
    for (int j = 0; j < 4; j++)
        *(vf4*)&Bs[0][rA + 64 * j][ak] = *(const vf4*)&gB[(size_t)j * rs64];
    __syncthreads();
#endif

    for (int step = 0; step < nsteps; ++step) {
        const int cur = step & 1;
        const int nxt = cur ^ 1;
        const bool more = (step + 1) < nsteps;

#if TDM_ARITY > 0
        // kick off the DMA for the next K-slab into the alternate buffer;
        // it runs concurrently with the WMMAs below.
        if (more && tid < 32) {
            const size_t ko = (size_t)(step + 1) * BK;
            tdm_load_tile(gAt + ko, aBase[nxt], K, BM);
            tdm_load_tile(gBt + ko, bBase[nxt], K, BNR);
        }
#else
        // fetch next K-slab into registers while WMMAs run on `cur`
        vf4 ra0, ra1, rb[4];
        if (more) {
            const size_t ko = (size_t)(step + 1) * BK;
            ra0 = *(const vf4*)&gA[ko];
            ra1 = *(const vf4*)&gA[rs64 + ko];
            #pragma unroll
            for (int j = 0; j < 4; j++)
                rb[j] = *(const vf4*)&gB[(size_t)j * rs64 + ko];
            if (step + 2 < nsteps) {
                __builtin_prefetch(&gA[ko + BK], 0, 1);
                __builtin_prefetch(&gB[ko + BK], 0, 1);
            }
        }
#endif

        // B fragments: 16-bit B 32x16 layout — lane half selects K {0..7,16..23}
        // vs {8..15,24..31}, col = lane%16 (BT row-contiguous)
        Frag bf[4];
        #pragma unroll
        for (int nt = 0; nt < 4; nt++) {
            const int col = wn * 64 + nt * 16 + lr;
            bf[nt].h[0] = *(const v8h*)&Bs[cur][col][hs * 8];
            bf[nt].h[1] = *(const v8h*)&Bs[cur][col][16 + hs * 8];
        }
        // A fragments: 16-bit A 16x32 layout — lane half selects K {0..7,16..23}
        // vs {8..15,24..31}, row = lane%16; one A frag live at a time.
        #pragma unroll
        for (int mt = 0; mt < 4; mt++) {
            Frag af;
            const int row = wm * 64 + mt * 16 + lr;
            af.h[0] = *(const v8h*)&As[cur][row][hs * 8];
            af.h[1] = *(const v8h*)&As[cur][row][16 + hs * 8];
            #pragma unroll
            for (int nt = 0; nt < 4; nt++)
                acc[mt][nt] = __builtin_amdgcn_wmma_f32_16x16x32_f16(
                    false, af.v, false, bf[nt].v,
                    (short)0, acc[mt][nt], false, false);
        }

#if TDM_ARITY > 0
        // issuing wave drains TENSORcnt, then the barrier publishes `nxt`
        if (tid < 32) __builtin_amdgcn_s_wait_tensorcnt(0);
        __syncthreads();
#else
        // publish next slab into the alternate buffer; one barrier per step
        if (more) {
            *(vf4*)&As[nxt][rA     ][ak] = ra0;
            *(vf4*)&As[nxt][rA + 64][ak] = ra1;
            #pragma unroll
            for (int j = 0; j < 4; j++)
                *(vf4*)&Bs[nxt][rA + 64 * j][ak] = rb[j];
        }
        __syncthreads();
#endif
    }

    // epilogue; C/D layout: VGPR r -> row = hs*8 + r, col = lane%16
    #pragma unroll
    for (int mt = 0; mt < 4; mt++) {
        #pragma unroll
        for (int nt = 0; nt < 4; nt++) {
            const int col = blockN + wn * 64 + nt * 16 + lr;
            #pragma unroll
            for (int r = 0; r < 8; r++) {
                const int row = blockM + wm * 64 + mt * 16 + hs * 8 + r;
                float vv = acc[mt][nt][r];
                if (mode == 1) {
                    vv += bias[col];
                    vv = vv > 0.f ? vv : 0.f;
                    outF16[(size_t)row * N + col] = (_Float16)vv;
                } else {
                    outF32[(size_t)row * N + col] = vv;
                }
            }
        }
    }
}

// ---------------------------------------------------------------------------
// Kernel 5: y = gemm2_out + b2 + x_residual ; LayerNorm2 -> final output
// ---------------------------------------------------------------------------
__global__ __launch_bounds__(128) void ln2_kernel(
    const float* __restrict__ Y, const float* __restrict__ b2,
    const float* __restrict__ X, const float* __restrict__ g,
    const float* __restrict__ b, float* __restrict__ out)
{
    const int token = blockIdx.x;
    const int tid   = threadIdx.x;   // 128 threads, 6 elems each
    __shared__ float s_a[128], s_b[128];

    float y[6];
    float lsum = 0.f, lsq = 0.f;
    #pragma unroll
    for (int j = 0; j < 6; j++) {
        const int idx = tid * 6 + j;
        y[j] = Y[(size_t)token * DM + idx] + b2[idx] + X[(size_t)token * DM + idx];
        lsum += y[j];
        lsq  += y[j] * y[j];
    }
    s_a[tid] = lsum; s_b[tid] = lsq;
    __syncthreads();
    for (int s = 64; s > 0; s >>= 1) {
        if (tid < s) { s_a[tid] += s_a[tid + s]; s_b[tid] += s_b[tid + s]; }
        __syncthreads();
    }
    const float mean = s_a[0] * (1.f / DM);
    const float var  = s_b[0] * (1.f / DM) - mean * mean;
    const float inv  = rsqrtf(var + 1e-6f);
    #pragma unroll
    for (int j = 0; j < 6; j++) {
        const int idx = tid * 6 + j;
        out[(size_t)token * DM + idx] = (y[j] - mean) * inv * g[idx] + b[idx];
    }
}

// ---------------------------------------------------------------------------
extern "C" void kernel_launch(void* const* d_in, const int* in_sizes, int n_in,
                              void* d_out, int out_size, void* d_ws, size_t ws_size,
                              hipStream_t stream)
{
    const float* enc  = (const float*)d_in[0];
    const float* mask = (const float*)d_in[1];
    const float* wq   = (const float*)d_in[2];
    const float* bq   = (const float*)d_in[3];
    const float* wk   = (const float*)d_in[4];
    const float* bk   = (const float*)d_in[5];
    const float* wv   = (const float*)d_in[6];
    const float* bv   = (const float*)d_in[7];
    const float* g1   = (const float*)d_in[8];
    const float* b1v  = (const float*)d_in[9];
    const float* w1   = (const float*)d_in[10];
    const float* b1   = (const float*)d_in[11];
    const float* w2   = (const float*)d_in[12];
    const float* b2   = (const float*)d_in[13];
    const float* g2   = (const float*)d_in[14];
    const float* b2v  = (const float*)d_in[15];
    float* out = (float*)d_out;
    (void)in_sizes; (void)n_in; (void)out_size; (void)ws_size;

    uint8_t* ws = (uint8_t*)d_ws;
    size_t off = 0;
    auto carve = [&](size_t bytes) -> void* {
        void* p = ws + off;
        off += (bytes + 255) & ~(size_t)255;
        return p;
    };
    float*    Xf32 = (float*)   carve((size_t)TOK * DM * sizeof(float));
    _Float16* Xf16 = (_Float16*)carve((size_t)TOK * DM * sizeof(_Float16));
    _Float16* W1T  = (_Float16*)carve((size_t)DI * DM * sizeof(_Float16));
    _Float16* W2T  = (_Float16*)carve((size_t)DM * DI * sizeof(_Float16));
    _Float16* Hf16 = (_Float16*)carve((size_t)TOK * DI * sizeof(_Float16));
    float*    Yf32 = (float*)   carve((size_t)TOK * DM * sizeof(float));

    // 1) attention + LN1
    attn_ln1_kernel<<<TOK, 128, 0, stream>>>(enc, mask, wq, bq, wk, bk, wv, bv,
                                             g1, b1v, Xf32, Xf16);

    // 2) weight transpose+convert: W1T[n*DM+k] = w1[k*DI+n]; W2T[n*DI+k] = w2[k*DM+n]
    {
        const size_t tot = (size_t)DM * DI;
        const int blk = 256;
        const int grd = (int)((tot + blk - 1) / blk);
        cvt_transpose_kernel<<<grd, blk, 0, stream>>>(w1, W1T, DM, DI);
        cvt_transpose_kernel<<<grd, blk, 0, stream>>>(w2, W2T, DI, DM);
    }

    // 3) GEMM1: H = relu(X * W1 + b1)  (8192 x 3072, K=768)
    {
        dim3 grid(DI / BNR, TOK / BM);
        wmma_gemm_kernel<<<grid, 256, 0, stream>>>(Xf16, W1T, nullptr, Hf16, b1,
                                                   TOK, DI, DM, 1);
    }
    // 4) GEMM2: Y = H * W2  (8192 x 768, K=3072), f32 out
    {
        dim3 grid(DM / BNR, TOK / BM);
        wmma_gemm_kernel<<<grid, 256, 0, stream>>>(Hf16, W2T, Yf32, nullptr, nullptr,
                                                   TOK, DM, DI, 0);
    }
    // 5) +b2 + residual, LN2
    ln2_kernel<<<TOK, 128, 0, stream>>>(Yf32, b2, Xf32, g2, b2v, out);
}
